// Attention_4853313044856
// MI455X (gfx1250) — matrix operand order repaired and checked by reading, not compiled, not run on hardware
//
#include <hip/hip_runtime.h>

typedef __bf16 bf16;
typedef __attribute__((ext_vector_type(16))) __bf16 v16bf;
typedef __attribute__((ext_vector_type(8)))  float  v8f;

#define WMMA_BF16(a, b, c) \
    __builtin_amdgcn_wmma_f32_16x16x32_bf16(false, (a), false, (b), (short)0, (c), false, false)

__device__ inline v8f v8f_zero() {
    v8f r;
#pragma unroll
    for (int i = 0; i < 8; ++i) r[i] = 0.0f;
    return r;
}

// ---------------------------------------------------------------------------
// Cross-lane butterfly reductions within 16-lane halves using v_permlane16_b32
// (pure VALU; no LDS round trips). Lane i reads lane i^m within its group.
// ---------------------------------------------------------------------------
__device__ inline float xor16_get(float v, int lo, int hi) {
    int x = __float_as_int(v);
    int p = __builtin_amdgcn_permlane16(x, x, lo, hi, false, false);
    return __int_as_float(p);
}
__device__ inline float red_max16(float v) {
    v = fmaxf(v, xor16_get(v, (int)0x67452301, (int)0xEFCDAB89));  // m=1
    v = fmaxf(v, xor16_get(v, (int)0x54761032, (int)0xDCFE98BA));  // m=2
    v = fmaxf(v, xor16_get(v, (int)0x32107654, (int)0xBA98FEDC));  // m=4
    v = fmaxf(v, xor16_get(v, (int)0xFEDCBA98, (int)0x76543210));  // m=8
    return v;
}
__device__ inline float red_sum16(float v) {
    v += xor16_get(v, (int)0x67452301, (int)0xEFCDAB89);
    v += xor16_get(v, (int)0x54761032, (int)0xDCFE98BA);
    v += xor16_get(v, (int)0x32107654, (int)0xBA98FEDC);
    v += xor16_get(v, (int)0xFEDCBA98, (int)0x76543210);
    return v;
}

// Load a 16x32 bf16 A-fragment: lane holds row, elements at k = half*8 + j
// (j<8) and 16 + half*8 + (j-8)  ->  two 16B loads per lane.
__device__ inline v16bf load_afrag(const bf16* rowbase, int half) {
    union { v16bf v; uint4 u[2]; } fa;
    fa.u[0] = *(const uint4*)(rowbase + half * 8);
    fa.u[1] = *(const uint4*)(rowbase + 16 + half * 8);
    return fa.v;
}

// Load a 32x16 bf16 B-fragment from K-contiguous memory: lane holds column,
// elements k = half*8 + j (j<8) and 16 + half*8 + (j-8)  ->  two 16B loads per lane.
__device__ inline v16bf load_bfrag(const bf16* colbase, int half) {
    union { v16bf v; uint4 u[2]; } fb;
    const bf16* p = colbase + half * 8;
    fb.u[0] = *(const uint4*)p;
    fb.u[1] = *(const uint4*)(p + 16);
    return fb.v;
}

// ---------------------------------------------------------------------------
// f32 -> bf16 conversion (vectorized, row-major preserved)
// ---------------------------------------------------------------------------
__global__ __launch_bounds__(256) void cvt_f32_bf16(const float* __restrict__ in,
                                                    bf16* __restrict__ out, int n) {
    int i = (blockIdx.x * blockDim.x + threadIdx.x) * 4;
    if (i + 4 <= n) {
        float4 f = *(const float4*)(in + i);
        out[i + 0] = (bf16)f.x;
        out[i + 1] = (bf16)f.y;
        out[i + 2] = (bf16)f.z;
        out[i + 3] = (bf16)f.w;
    } else {
        for (; i < n; ++i) out[i] = (bf16)in[i];
    }
}

// ---------------------------------------------------------------------------
// f32 -> bf16 convert + transpose: in[R][C] (f32) -> out[C][R] (bf16)
// ---------------------------------------------------------------------------
__global__ __launch_bounds__(256) void cvt_transpose_f32_bf16(const float* __restrict__ in,
                                                              bf16* __restrict__ out,
                                                              int R, int C) {
    __shared__ float tile[32][33];
    int bx = blockIdx.x * 32;
    int by = blockIdx.y * 32;
    int tx = threadIdx.x;
#pragma unroll
    for (int j = threadIdx.y; j < 32; j += 8)
        tile[j][tx] = in[(size_t)(by + j) * C + bx + tx];
    __syncthreads();
#pragma unroll
    for (int j = threadIdx.y; j < 32; j += 8)
        out[(size_t)(bx + j) * R + by + tx] = (bf16)tile[tx][j];
}

// ---------------------------------------------------------------------------
// LDS-free bf16 WMMA GEMM:  C[M,N] = A[M,K] * Bt[N,K]^T
// block = 128 threads (4 waves); block tile 128x128; wave tile 64x64
// -> 16 wmma per K-step against 16 global b128 loads (1:1).
// MODE: 0 = f32 row-major out, 1 = bf16 row-major out, 2 = bf16 transposed out
// ---------------------------------------------------------------------------
template <int MODE>
__global__ __launch_bounds__(128) void gemm_tile(const bf16* __restrict__ A,
                                                 const bf16* __restrict__ Bt,
                                                 void* __restrict__ Cout,
                                                 int M, int N, int K) {
    const int tid  = threadIdx.x;
    const int wave = tid >> 5;
    const int lane = tid & 31;
    const int half = lane >> 4;
    const int rl   = lane & 15;

    const int bm = blockIdx.y * 128 + (wave >> 1) * 64;
    const int bn = blockIdx.x * 128 + (wave & 1) * 64;

    v8f acc[4][4];
#pragma unroll
    for (int i = 0; i < 4; ++i)
#pragma unroll
        for (int t = 0; t < 4; ++t) acc[i][t] = v8f_zero();

    const bf16* arow[4];
#pragma unroll
    for (int i = 0; i < 4; ++i) arow[i] = A + (size_t)(bm + i * 16 + rl) * K;
    const bf16* brow[4];
#pragma unroll
    for (int t = 0; t < 4; ++t) brow[t] = Bt + (size_t)(bn + t * 16 + rl) * K;

    for (int k0 = 0; k0 < K; k0 += 32) {
        v16bf af[4];
#pragma unroll
        for (int i = 0; i < 4; ++i) af[i] = load_afrag(arow[i] + k0, half);
#pragma unroll
        for (int t = 0; t < 4; ++t) {
            v16bf b = load_bfrag(brow[t] + k0, half);
#pragma unroll
            for (int i = 0; i < 4; ++i) acc[i][t] = WMMA_BF16(af[i], b, acc[i][t]);
        }
    }

    // Store C (C layout: lane<16 rows +0..7, lane>=16 rows +8..15, col = rl)
#pragma unroll
    for (int i = 0; i < 4; ++i)
#pragma unroll
        for (int t = 0; t < 4; ++t) {
            if (MODE == 2) {
                // transposed out[col][row]: the 8 e-values are consecutive rows
                // -> pack into one 16B store
                union { uint4 u; bf16 e[8]; } pk;
#pragma unroll
                for (int e = 0; e < 8; ++e) pk.e[e] = (bf16)acc[i][t][e];
                int col = bn + t * 16 + rl;
                int row = bm + i * 16 + half * 8;
                *(uint4*)((bf16*)Cout + (size_t)col * M + row) = pk.u;
            } else {
#pragma unroll
                for (int e = 0; e < 8; ++e) {
                    int row = bm + i * 16 + e + half * 8;
                    int col = bn + t * 16 + rl;
                    if (MODE == 0)
                        ((float*)Cout)[(size_t)row * N + col] = acc[i][t][e];
                    else
                        ((bf16*)Cout)[(size_t)row * N + col] = (bf16)acc[i][t][e];
                }
            }
        }
}

// ---------------------------------------------------------------------------
// RoPE on Q and K in place. Layout [s][h*128 + d], HD=128, 64 (cos,sin) pairs.
// ---------------------------------------------------------------------------
__global__ __launch_bounds__(256) void rope_kernel(bf16* __restrict__ Q, bf16* __restrict__ K,
                                                   const float* __restrict__ cs,
                                                   const float* __restrict__ sn) {
    int idx = blockIdx.x * blockDim.x + threadIdx.x;  // over S * D/2
    if (idx >= 2048 * 2048) return;
    int s = idx >> 11;
    int p = idx & 2047;
    int j = p & 63;
    float c  = cs[s * 64 + j];
    float si = sn[s * 64 + j];
    size_t base = (size_t)s * 4096 + (p << 1);
    float q0 = (float)Q[base], q1 = (float)Q[base + 1];
    Q[base]     = (bf16)(q0 * c - q1 * si);
    Q[base + 1] = (bf16)(q0 * si + q1 * c);
    float k0 = (float)K[base], k1 = (float)K[base + 1];
    K[base]     = (bf16)(k0 * c - k1 * si);
    K[base + 1] = (bf16)(k0 * si + k1 * c);
}

// ---------------------------------------------------------------------------
// Causal flash attention, barrier-free. grid = (S/64, H), block = 128 (4 waves).
// Each wave owns 16 q rows, loops its own causal KV range in blocks of 64.
// Softmax runs in the exp2 domain (1/sqrt(HD) * log2(e) folded into scores),
// saving one v_mul per exponential; invariant under the max-shift algebra.
// ---------------------------------------------------------------------------
__global__ __launch_bounds__(128) void flash_attn(const bf16* __restrict__ Q,
                                                  const bf16* __restrict__ Kt,
                                                  const bf16* __restrict__ Vt,
                                                  bf16* __restrict__ O) {
    const int Dm = 4096, HD = 128, S = 2048;
    const float SCALE_L2E = 0.12750285252539663f;  // (1/sqrt(128)) * log2(e)

    const int h    = blockIdx.y;
    const int tid  = threadIdx.x;
    const int wave = tid >> 5;
    const int lane = tid & 31;
    const int half = lane >> 4;
    const int rl   = lane & 15;
    const int qb   = blockIdx.x * 64 + wave * 16;

    __shared__ bf16 Ps[4][16 * 72];  // per-wave 16x64 P scratch, row stride 72

    // Persistent Q fragments (A-matrix layout), 4 k-steps over HD=128
    v16bf qf[4];
    const bf16* qrow = Q + (size_t)(qb + rl) * Dm + h * HD;
#pragma unroll
    for (int f = 0; f < 4; ++f) qf[f] = load_afrag(qrow + f * 32, half);

    float m[8], l[8];
    v8f o[8];
#pragma unroll
    for (int e = 0; e < 8; ++e) { m[e] = -3.0e38f; l[e] = 0.0f; }
#pragma unroll
    for (int n = 0; n < 8; ++n) o[n] = v8f_zero();

    const int kb_end = qb + 16;  // exact causal bound for this wave's rows
    for (int kb = 0; kb < kb_end; kb += 64) {
        // ---- S = Q * K^T : four 16x16 tiles (64 keys) ----
        v8f s[4];
#pragma unroll
        for (int t = 0; t < 4; ++t) {
            v8f acc = v8f_zero();
            const bf16* kcol = Kt + (size_t)(kb + t * 16 + rl) * Dm + h * HD;
#pragma unroll
            for (int f = 0; f < 4; ++f)
                acc = WMMA_BF16(qf[f], load_bfrag(kcol + f * 32, half), acc);
            s[t] = acc;
        }

        // ---- online softmax, exp2 domain ----
        float alpha[8];
#pragma unroll
        for (int e = 0; e < 8; ++e) {
            int qrow_i = qb + e + half * 8;
            float rmax = -3.0e38f;
#pragma unroll
            for (int t = 0; t < 4; ++t) {
                int key = kb + t * 16 + rl;
                float val = s[t][e] * SCALE_L2E + (key > qrow_i ? -1.0e9f : 0.0f);
                s[t][e] = val;
                rmax = fmaxf(rmax, val);
            }
            rmax = red_max16(rmax);
            float mnew = fmaxf(m[e], rmax);
            alpha[e] = exp2f(m[e] - mnew);
            m[e] = mnew;
            float rsum = 0.0f;
#pragma unroll
            for (int t = 0; t < 4; ++t) {
                float p = exp2f(s[t][e] - mnew);
                s[t][e] = p;
                rsum += p;
            }
            rsum = red_sum16(rsum);
            l[e] = l[e] * alpha[e] + rsum;
        }
#pragma unroll
        for (int n = 0; n < 8; ++n)
#pragma unroll
            for (int e = 0; e < 8; ++e) o[n][e] *= alpha[e];

        // ---- P (bf16) via per-wave LDS round trip into A-frag layout ----
#pragma unroll
        for (int t = 0; t < 4; ++t)
#pragma unroll
            for (int e = 0; e < 8; ++e)
                Ps[wave][(e + half * 8) * 72 + t * 16 + rl] = (bf16)s[t][e];
        asm volatile("s_wait_dscnt 0x0" ::: "memory");  // wave-local LDS ordering

        v16bf pa[2];  // two 16x32 A-frags covering keys kb..+31, kb+32..+63
        {
            const unsigned int* pp = (const unsigned int*)(Ps[wave] + rl * 72);
#pragma unroll
            for (int f = 0; f < 2; ++f) {
                union { v16bf v; unsigned int u[8]; } fa;
#pragma unroll
                for (int jj = 0; jj < 4; ++jj) fa.u[jj]     = pp[f * 16 + half * 4 + jj];
#pragma unroll
                for (int jj = 0; jj < 4; ++jj) fa.u[4 + jj] = pp[f * 16 + 8 + half * 4 + jj];
                pa[f] = fa.v;
            }
        }

        // ---- O += P * V : 8 d-tiles x 2 key k-steps, V^T rows contiguous ----
#pragma unroll
        for (int f = 0; f < 2; ++f)
#pragma unroll
            for (int n = 0; n < 8; ++n) {
                const bf16* vrow = Vt + (size_t)(h * HD + n * 16 + rl) * S + kb + f * 32;
                o[n] = WMMA_BF16(pa[f], load_bfrag(vrow, half), o[n]);
            }
    }

    // ---- epilogue: normalize (explicit v_rcp + mul) and store bf16 ----
    float linv[8];
#pragma unroll
    for (int e = 0; e < 8; ++e) linv[e] = __builtin_amdgcn_rcpf(l[e]);
#pragma unroll
    for (int n = 0; n < 8; ++n)
#pragma unroll
        for (int e = 0; e < 8; ++e) {
            int qrow_i = qb + e + half * 8;
            int col    = h * HD + n * 16 + rl;
            O[(size_t)qrow_i * Dm + col] = (bf16)(o[n][e] * linv[e]);
        }
}

// ---------------------------------------------------------------------------
// Host-side launch
// ---------------------------------------------------------------------------
extern "C" void kernel_launch(void* const* d_in, const int* in_sizes, int n_in,
                              void* d_out, int out_size, void* d_ws, size_t ws_size,
                              hipStream_t stream) {
    const int S = 2048, D = 4096, H = 32;

    const float* x  = (const float*)d_in[0];
    const float* wq = (const float*)d_in[1];
    const float* wk = (const float*)d_in[2];
    const float* wv = (const float*)d_in[3];
    const float* wo = (const float*)d_in[4];
    const float* fc = (const float*)d_in[5];
    const float* fs = (const float*)d_in[6];
    // d_in[7] = mask, applied analytically (causal) inside flash_attn
    float* out = (float*)d_out;

    // Workspace layout (bf16): x | wq^T | wk^T | wv^T | wo^T | Q | K | V^T | attn_out
    bf16* xb  = (bf16*)d_ws;
    bf16* wqt = xb  + (size_t)S * D;
    bf16* wkt = wqt + (size_t)D * D;
    bf16* wvt = wkt + (size_t)D * D;
    bf16* wot = wvt + (size_t)D * D;
    bf16* Qb  = wot + (size_t)D * D;
    bf16* Kb  = Qb  + (size_t)S * D;
    bf16* Vt  = Kb  + (size_t)S * D;   // [D][S]
    bf16* Ob  = Vt  + (size_t)S * D;

    // 1) Convert x (row-major) and weights (transposed) to bf16.
    {
        int nx = S * D;
        cvt_f32_bf16<<<(nx / 4 + 255) / 256, 256, 0, stream>>>(x, xb, nx);
        dim3 gt(D / 32, D / 32), bt(32, 8);
        cvt_transpose_f32_bf16<<<gt, bt, 0, stream>>>(wq, wqt, D, D);
        cvt_transpose_f32_bf16<<<gt, bt, 0, stream>>>(wk, wkt, D, D);
        cvt_transpose_f32_bf16<<<gt, bt, 0, stream>>>(wv, wvt, D, D);
        cvt_transpose_f32_bf16<<<gt, bt, 0, stream>>>(wo, wot, D, D);
    }

    // 2) QKV projections (LDS-free WMMA GEMM). V written transposed ([D][S]).
    dim3 g1(D / 128, S / 128);
    gemm_tile<1><<<g1, 128, 0, stream>>>(xb, wqt, Qb, S, D, D);
    gemm_tile<1><<<g1, 128, 0, stream>>>(xb, wkt, Kb, S, D, D);
    gemm_tile<2><<<g1, 128, 0, stream>>>(xb, wvt, Vt, S, D, D);

    // 3) RoPE on Q and K
    {
        int n = S * D / 2;
        rope_kernel<<<(n + 255) / 256, 256, 0, stream>>>(Qb, Kb, fc, fs);
    }

    // 4) Causal flash attention -> Ob ([s][h*HD+d] bf16), barrier-free
    dim3 g2(S / 64, H);
    flash_attn<<<g2, 128, 0, stream>>>(Qb, Kb, Vt, Ob);

    // 5) Output projection (f32 out to d_out)
    gemm_tile<0><<<g1, 128, 0, stream>>>(Ob, wot, out, S, D, D);
}
